// NMP_22832046146013
// MI455X (gfx1250) — matrix operand order repaired and checked by reading, not compiled, not run on hardware
//
#include <hip/hip_runtime.h>
#include <hip/hip_bf16.h>

typedef __attribute__((ext_vector_type(16))) __bf16 v16bf;
typedef __attribute__((ext_vector_type(8)))  float  v8f;

union AFrag {
    v16bf v;
    unsigned u[8];
};
union BFrag {
    v16bf v;
    uint4 q[2];
};
union BfPair {
    unsigned u;
    __bf16 h[2];
};

__device__ __forceinline__ unsigned pack_bf16x2(float lo, float hi) {
    BfPair p;
    p.h[0] = (__bf16)lo;
    p.h[1] = (__bf16)hi;
    return p.u;
}

// ---------------------------------------------------------------------------
// Pack W [32 x 4096] (k-major, inner = i*64 + o) and bias [4096] into bf16
// B-fragments per lane in the contiguous k order (k = 16h + s) that the A operand also uses:
//   dword index = kb*1024 + nt*256 + lane*8 + d
//   lane: column n = nt*16 + (lane&15); K-base = (lane<16 ? 0 : 16)
//   dword d holds K = Kbase+2d (lo16), Kbase+2d+1 (hi16)
// kb in [0,64): B[kk,n] = W[kk, kb*64 + n]   (kb == i index)
// kb in [64,66): bias tail, B[kk,n] = bias[((kb-64)*32+kk)*64 + n]
// ---------------------------------------------------------------------------
__global__ void pack_kernel(const float* __restrict__ W,
                            const float* __restrict__ bias,
                            unsigned* __restrict__ Bp) {
    int t = blockIdx.x * 256 + threadIdx.x;
    if (t >= 66 * 1024) return;
    int d    = t & 7;
    int lane = (t >> 3) & 31;
    int nt   = (t >> 8) & 3;
    int kb   = t >> 10;
    int n     = nt * 16 + (lane & 15);
    int kbase = (lane < 16) ? 0 : 16;
    int j0    = kbase + 2 * d;
    float v0, v1;
    if (kb < 64) {
        v0 = W[(size_t)j0 * 4096 + kb * 64 + n];
        v1 = W[(size_t)(j0 + 1) * 4096 + kb * 64 + n];
    } else {
        int i0 = (kb - 64) * 32 + j0;
        v0 = bias[i0 * 64 + n];
        v1 = bias[(i0 + 1) * 64 + n];
    }
    Bp[t] = pack_bf16x2(v0, v1);
}

__global__ void zero_kernel(float* __restrict__ p, int n) {
    int t = blockIdx.x * 256 + threadIdx.x;
    if (t < n) p[t] = 0.0f;
}

// ---------------------------------------------------------------------------
// Fused NNConv message+scatter. One wave handles 32 edges (two 16-row M-tiles)
// x 64 outputs, so every B fragment loaded feeds TWO WMMAs (2x arithmetic
// intensity vs one M-tile).
//   msg = (outer(h_src, efeat) | h_src) @ Bpack   via 66 K-steps of WMMA bf16
// then atomic scatter-add into sums[dst], counts into cnt[dst].
// ---------------------------------------------------------------------------
#define WAVES 4
__global__ __launch_bounds__(128) void edge_conv_kernel(
    const float* __restrict__ hfeat,   // [n_src, 64]
    const float* __restrict__ efeat,   // [E, 32]
    const unsigned* __restrict__ Bp,   // packed bf16 B fragments
    const int* __restrict__ src,
    const int* __restrict__ dst,
    int E,
    float* __restrict__ sums,          // [n_dst, 64]
    float* __restrict__ cnt) {         // [n_dst]
    __shared__ __align__(16) float sh_h [WAVES][32][68]; // stride 68: 16B-aligned, conflict-free
    __shared__ __align__(16) float sh_ef[WAVES][32][36]; // stride 36: 16B-aligned, conflict-free

    const int lane = threadIdx.x & 31;
    const int w    = threadIdx.x >> 5;
    const int tile = blockIdx.x * WAVES + w;
    const int ebase = tile * 32;

    // Stage h[src] rows (32 x 64 f32) into LDS, zero for out-of-range edges.
    for (int idx = lane; idx < 512; idx += 32) {
        int e = idx >> 4, c = idx & 15;
        float4 v = make_float4(0.f, 0.f, 0.f, 0.f);
        int ge = ebase + e;
        if (ge < E) {
            int s = src[ge];
            v = *(const float4*)(hfeat + (size_t)s * 64 + c * 4);
        }
        *(float4*)&sh_h[w][e][c * 4] = v;
    }
    // Stage efeat rows (32 x 32 f32).
    for (int idx = lane; idx < 256; idx += 32) {
        int e = idx >> 3, c = idx & 7;
        float4 v = make_float4(0.f, 0.f, 0.f, 0.f);
        int ge = ebase + e;
        if (ge < E) v = *(const float4*)(efeat + (size_t)ge * 32 + c * 4);
        *(float4*)&sh_ef[w][e][c * 4] = v;
    }
    __syncthreads();

    const int e   = lane & 15;
    const int hi  = lane >> 4;
    const int kA0 = hi ? 16 : 0;   // A k order used here: lanes 0-15 K{0..15}
    const int kA1 = hi ? 24 : 8;   //                  lanes16-31 K{16..31}
    const float* h0 = &sh_h[w][e][0];
    const float* h1 = &sh_h[w][e + 16][0];
    const float* e0 = &sh_ef[w][e][0];
    const float* e1 = &sh_ef[w][e + 16][0];

    v8f acc[2][4];
    #pragma unroll
    for (int mt = 0; mt < 2; ++mt)
        #pragma unroll
        for (int nt = 0; nt < 4; ++nt) {
            v8f z = {0.f, 0.f, 0.f, 0.f, 0.f, 0.f, 0.f, 0.f};
            acc[mt][nt] = z;
        }

    // Main contraction: kb == i index; A[e,kk] = h[e,i] * efeat[e,kk]
    for (int kb = 0; kb < 64; ++kb) {
        float s0 = h0[kb];
        float s1 = h1[kb];
        AFrag a0, a1;
        #pragma unroll
        for (int j = 0; j < 8; ++j) {
            a0.v[j]     = (__bf16)(s0 * e0[kA0 + j]);
            a0.v[8 + j] = (__bf16)(s0 * e0[kA1 + j]);
            a1.v[j]     = (__bf16)(s1 * e1[kA0 + j]);
            a1.v[8 + j] = (__bf16)(s1 * e1[kA1 + j]);
        }
        size_t bofs = (size_t)kb * 1024 + (size_t)lane * 8;
        #pragma unroll
        for (int nt = 0; nt < 4; ++nt) {
            BFrag b;
            const uint4* p = (const uint4*)(Bp + bofs + nt * 256);
            b.q[0] = p[0];
            b.q[1] = p[1];
            acc[0][nt] = __builtin_amdgcn_wmma_f32_16x16x32_bf16(
                false, a0.v, false, b.v, (short)0, acc[0][nt], false, false);
            acc[1][nt] = __builtin_amdgcn_wmma_f32_16x16x32_bf16(
                false, a1.v, false, b.v, (short)0, acc[1][nt], false, false);
        }
    }
    // Bias tail: A[e,kk] = h[e, t*32 + kk]
    #pragma unroll
    for (int t = 0; t < 2; ++t) {
        const float* v0 = h0 + t * 32;
        const float* v1 = h1 + t * 32;
        AFrag a0, a1;
        #pragma unroll
        for (int j = 0; j < 8; ++j) {
            a0.v[j]     = (__bf16)v0[kA0 + j];
            a0.v[8 + j] = (__bf16)v0[kA1 + j];
            a1.v[j]     = (__bf16)v1[kA0 + j];
            a1.v[8 + j] = (__bf16)v1[kA1 + j];
        }
        size_t bofs = (size_t)(64 + t) * 1024 + (size_t)lane * 8;
        #pragma unroll
        for (int nt = 0; nt < 4; ++nt) {
            BFrag b;
            const uint4* p = (const uint4*)(Bp + bofs + nt * 256);
            b.q[0] = p[0];
            b.q[1] = p[1];
            acc[0][nt] = __builtin_amdgcn_wmma_f32_16x16x32_bf16(
                false, a0.v, false, b.v, (short)0, acc[0][nt], false, false);
            acc[1][nt] = __builtin_amdgcn_wmma_f32_16x16x32_bf16(
                false, a1.v, false, b.v, (short)0, acc[1][nt], false, false);
        }
    }

    // Scatter: C layout -> VGPR r holds rows {r, r+8}, lane&15 = column.
    const int col = lane & 15;
    #pragma unroll
    for (int mt = 0; mt < 2; ++mt) {
        #pragma unroll
        for (int r = 0; r < 8; ++r) {
            int ge = ebase + mt * 16 + r + (hi ? 8 : 0);
            if (ge < E) {
                int dd = dst[ge];
                float* orow = sums + (size_t)dd * 64 + col;
                atomicAdd(orow + 0,  acc[mt][0][r]);
                atomicAdd(orow + 16, acc[mt][1][r]);
                atomicAdd(orow + 32, acc[mt][2][r]);
                atomicAdd(orow + 48, acc[mt][3][r]);
            }
        }
    }
    // Counts: one lane per edge (32 edges, 32 lanes).
    {
        int ge = ebase + lane;
        if (ge < E) atomicAdd(cnt + dst[ge], 1.0f);
    }
}

__global__ void finalize_kernel(const float* __restrict__ sums,
                                const float* __restrict__ cnt,
                                float* __restrict__ out,
                                int total, int do_relu) {
    int t = blockIdx.x * 256 + threadIdx.x;
    if (t >= total) return;
    int node = t >> 6;
    float c = cnt[node];
    c = (c < 1.0f) ? 1.0f : c;
    float v = sums[t] / c;
    if (do_relu) v = (v > 0.0f) ? v : 0.0f;
    out[t] = v;
}

extern "C" void kernel_launch(void* const* d_in, const int* in_sizes, int n_in,
                              void* d_out, int out_size, void* d_ws, size_t ws_size,
                              hipStream_t stream) {
    const float* in_feat = (const float*)d_in[0];
    const float* ef1     = (const float*)d_in[1];
    const float* ef2     = (const float*)d_in[2];
    const float* W1      = (const float*)d_in[3];
    const float* b1      = (const float*)d_in[4];
    const float* W2      = (const float*)d_in[5];
    const float* b2      = (const float*)d_in[6];
    const int*   src1    = (const int*)d_in[7];
    const int*   dst1    = (const int*)d_in[8];
    const int*   src2    = (const int*)d_in[9];
    const int*   dst2    = (const int*)d_in[10];

    const int E1   = in_sizes[7];
    const int E2   = in_sizes[9];
    const int NMID = 20000;
    const int NOUT = out_size / 64;

    const int PACKN = 66 * 1024;          // dwords per packed weight set
    unsigned* Bp1  = (unsigned*)d_ws;
    unsigned* Bp2  = Bp1 + PACKN;
    float* sums1   = (float*)(Bp2 + PACKN);
    float* cnt1    = sums1 + (size_t)NMID * 64;
    float* sums2   = cnt1 + NMID;
    float* cnt2    = sums2 + (size_t)NOUT * 64;
    int zeroN      = NMID * 64 + NMID + NOUT * 64 + NOUT;

    // 1) pack weights+bias to bf16 WMMA fragment layout
    pack_kernel<<<(PACKN + 255) / 256, 256, 0, stream>>>(W1, b1, Bp1);
    pack_kernel<<<(PACKN + 255) / 256, 256, 0, stream>>>(W2, b2, Bp2);
    // 2) zero accumulators (graph replays reuse ws)
    zero_kernel<<<(zeroN + 255) / 256, 256, 0, stream>>>(sums1, zeroN);
    // 3) layer 1: fused message + scatter (32 edges per wave)
    int tiles1 = (E1 + 31) / 32;
    edge_conv_kernel<<<(tiles1 + WAVES - 1) / WAVES, 128, 0, stream>>>(
        in_feat, ef1, Bp1, src1, dst1, E1, sums1, cnt1);
    // 4) mean + relu (in place -> h_mid)
    finalize_kernel<<<(NMID * 64 + 255) / 256, 256, 0, stream>>>(
        sums1, cnt1, sums1, NMID * 64, 1);
    // 5) layer 2
    int tiles2 = (E2 + 31) / 32;
    edge_conv_kernel<<<(tiles2 + WAVES - 1) / WAVES, 128, 0, stream>>>(
        sums1, ef2, Bp2, src2, dst2, E2, sums2, cnt2);
    // 6) mean -> output
    finalize_kernel<<<(NOUT * 64 + 255) / 256, 256, 0, stream>>>(
        sums2, cnt2, (float*)d_out, NOUT * 64, 0);
}
